// MiniMaxM2DecoderLayer_66571993088733
// MI455X (gfx1250) — matrix operand order repaired and checked by reading, not compiled, not run on hardware
//
#include <hip/hip_runtime.h>
#include <hip/hip_bf16.h>
#include <math.h>

// ---- problem constants -----------------------------------------------------
#define T_   2048          // tokens (B*S)
#define H_   2048          // hidden
#define NH_  16
#define NKV_ 4
#define HD_  128
#define RD_  64
#define DQ   2048          // NH*HD
#define DKV  512           // NKV*HD
#define E_   8
#define I_   1024
#define EPS_ 1e-6f
#define SCALE_ 0.08838834764831845f   // 128^-0.5

typedef __attribute__((ext_vector_type(16))) _Float16 v16h;
typedef __attribute__((ext_vector_type(8)))  _Float16 v8h;
typedef __attribute__((ext_vector_type(8)))  float    v8f;

__device__ __forceinline__ v8f zero8() {
  v8f z;
#pragma unroll
  for (int i = 0; i < 8; ++i) z[i] = 0.f;
  return z;
}

// A fragment (16x32 f16): lane<16 -> row M=lane, K {0..7,16..23};
// lane>=16 -> same rows, K {8..15,24..31}. rowp = this lane's row at k-chunk 0.
__device__ __forceinline__ v16h ld_a(const _Float16* rowp, int lane) {
  int hf = lane >> 4;
  const _Float16* p = rowp + hf * 8;
  v8h lo = *(const v8h*)(p);
  v8h hi = *(const v8h*)(p + 16);
  v16h a;
#pragma unroll
  for (int i = 0; i < 8; ++i) { a[i] = lo[i]; a[i + 8] = hi[i]; }
  return a;
}
__device__ __forceinline__ v16h ld_a_rm(const _Float16* base, int ld, int lane) {
  return ld_a(base + (size_t)(lane & 15) * ld, lane);
}
// B fragment (32x16 f16) from BT row-major [N][K]: lane<16 -> col N=lane, K {0..7,16..23};
// lane>=16 -> col N=lane-16, K {8..15,24..31}.
__device__ __forceinline__ v16h ld_b(const _Float16* base, int ld, int lane) {
  int l = lane & 15, hf = lane >> 4;
  const _Float16* p = base + (size_t)l * ld + hf * 8;
  v8h lo = *(const v8h*)(p);
  v8h hi = *(const v8h*)(p + 16);
  v16h b;
#pragma unroll
  for (int i = 0; i < 8; ++i) { b[i] = lo[i]; b[i + 8] = hi[i]; }
  return b;
}
__device__ __forceinline__ v8f wmma_f16(v16h a, v16h b, v8f c) {
  return __builtin_amdgcn_wmma_f32_16x16x32_f16(false, a, false, b, (short)0, c,
                                                false, false);
}
__device__ __forceinline__ float rmax16(float v) {
#pragma unroll
  for (int m = 8; m >= 1; m >>= 1) v = fmaxf(v, __shfl_xor(v, m, 32));
  return v;
}
__device__ __forceinline__ float rsum16(float v) {
#pragma unroll
  for (int m = 8; m >= 1; m >>= 1) v += __shfl_xor(v, m, 32);
  return v;
}
// LDS byte offset of a __shared__ object: LDS aperture keeps offset in addr[31:0]
__device__ __forceinline__ unsigned lds_off(const void* p) {
  return (unsigned)(size_t)p;
}
// async global -> LDS, 16B per lane (ASYNCcnt-tracked CDNA5 copy engine)
__device__ __forceinline__ void async_b128(unsigned ldsaddr, const void* gaddr) {
  asm volatile("global_load_async_to_lds_b128 %0, %1, off"
               :: "v"(ldsaddr), "v"(gaddr) : "memory");
}
__device__ __forceinline__ void wait_async0() {
  asm volatile("s_wait_asynccnt 0" ::: "memory");
}

// ---- weight convert + transpose: fp32 W[b][K][N] -> f16 WT[b][N][K] --------
__global__ void tr_f32_f16_kernel(const float* __restrict__ W,
                                  _Float16* __restrict__ WT, int K, int N) {
  __shared__ float tile[32][33];
  int b = blockIdx.z;
  int tx = threadIdx.x, ty = threadIdx.y;
  int n0 = blockIdx.x * 32, k0 = blockIdx.y * 32;
  const float* Wb = W + (size_t)b * K * N;
  _Float16* WTb = WT + (size_t)b * K * N;
#pragma unroll
  for (int i = 0; i < 32; i += 8)
    tile[ty + i][tx] = Wb[(size_t)(k0 + ty + i) * N + n0 + tx];
  __syncthreads();
#pragma unroll
  for (int i = 0; i < 32; i += 8)
    WTb[(size_t)(n0 + ty + i) * K + k0 + tx] = (_Float16)tile[tx][ty + i];
}

// ---- rmsnorm (fp32 in -> f16 out) -----------------------------------------
__global__ void rmsnorm_kernel(const float* __restrict__ X,
                               const float* __restrict__ W,
                               _Float16* __restrict__ OUT) {
  int t = blockIdx.x, tid = threadIdx.x;
  const float* row = X + (size_t)t * H_;
  __shared__ float red[256];
  float s = 0.f;
  for (int i = tid; i < H_; i += 256) { float v = row[i]; s += v * v; }
  red[tid] = s; __syncthreads();
  for (int o = 128; o > 0; o >>= 1) { if (tid < o) red[tid] += red[tid + o]; __syncthreads(); }
  float rs = rsqrtf(red[0] / H_ + EPS_);
  for (int i = tid; i < H_; i += 256)
    OUT[(size_t)t * H_ + i] = (_Float16)(row[i] * rs * W[i]);
}

// ---- QK-norm + partial RoPE (fp32 in -> f16 out, optional score prescale) --
__global__ void qknorm_rope_kernel(const float* __restrict__ RAW,
                                   const float* __restrict__ W,
                                   const float* __restrict__ COS,
                                   const float* __restrict__ SIN,
                                   _Float16* __restrict__ OUT, int D, float scale) {
  int t = blockIdx.x, tid = threadIdx.x;
  const float* row = RAW + (size_t)t * D;
  __shared__ float red[256];
  float s = 0.f;
  for (int i = tid; i < D; i += 256) { float v = row[i]; s += v * v; }
  red[tid] = s; __syncthreads();
  for (int o = 128; o > 0; o >>= 1) { if (tid < o) red[tid] += red[tid + o]; __syncthreads(); }
  float rs = rsqrtf(red[0] / D + EPS_);
  for (int i = tid; i < D; i += 256) {
    int d = i & (HD_ - 1);
    float v = row[i] * rs * W[i];
    float ov;
    if (d < RD_) {
      float c = COS[(size_t)t * RD_ + d], sn = SIN[(size_t)t * RD_ + d];
      if (d < RD_ / 2) { float p = row[i + RD_ / 2] * rs * W[i + RD_ / 2]; ov = v * c - p * sn; }
      else             { float p = row[i - RD_ / 2] * rs * W[i - RD_ / 2]; ov = v * c + p * sn; }
    } else ov = v;
    OUT[(size_t)t * D + i] = (_Float16)(ov * scale);
  }
}

// ---- main WMMA GEMM: C[M,N] = A[M,K](f16) * BT[N,K](f16) (+add) -----------
// 256 thr = 8 waves (2x4), wave tile 32x32 -> block tile 64x128.
// Async-LDS double-buffered pipeline: per 32-wide k-step stage A(64x32) +
// B(128x32) tiles via global_load_async_to_lds_b128, overlap with WMMAs.
#define LDSPAD 40   // halves per LDS row (80B: 16B-aligned, bank-spread)

__device__ __forceinline__ void store_tile(v8f c, int row0, int col0,
                                           const float* add, float* C,
                                           _Float16* C16, int ldc, int lane) {
  int l = lane & 15, hf = lane >> 4;
#pragma unroll
  for (int r = 0; r < 8; ++r) {
    size_t idx = (size_t)(row0 + r + hf * 8) * ldc + col0 + l;
    float v = c[r];
    if (add) v += add[idx];
    if (C)   C[idx] = v;
    if (C16) C16[idx] = (_Float16)v;
  }
}

__global__ void gemm_f16_kernel(const _Float16* __restrict__ A, int lda,
                                const _Float16* __restrict__ BT, int ldb,
                                const float* __restrict__ addsrc,
                                float* __restrict__ C, _Float16* __restrict__ C16,
                                int ldc, int K) {
  __shared__ _Float16 sA[2][64 * LDSPAD];
  __shared__ _Float16 sB[2][128 * LDSPAD];
  int tid = threadIdx.x;
  int lane = tid & 31, wv = tid >> 5;
  int rowblk = blockIdx.y * 64, colblk = blockIdx.x * 128;
  int lrow = (wv >> 2) * 32;   // wave-local row base in block tile
  int lcol = (wv & 3) * 32;    // wave-local col base in block tile

  // per-thread copy slots: A = 1x16B, B = 2x16B per k-step
  int arow = tid >> 2, aseg = tid & 3;     // 64 rows x 4 segs
  const _Float16* agsrc = A + (size_t)(rowblk + arow) * lda + aseg * 8;
  unsigned alds = aseg * 8;
  int br0 = tid >> 2, bs0 = tid & 3;       // rows 0..63
  int br1 = (tid + 256) >> 2, bs1 = tid & 3; // rows 64..127
  const _Float16* bgsrc0 = BT + (size_t)(colblk + br0) * ldb + bs0 * 8;
  const _Float16* bgsrc1 = BT + (size_t)(colblk + br1) * ldb + bs1 * 8;

  v8f c00 = zero8(), c01 = zero8(), c10 = zero8(), c11 = zero8();

  // prologue: stage k=0 into buffer 0
  async_b128(lds_off(&sA[0][arow * LDSPAD]) + alds, agsrc);
  async_b128(lds_off(&sB[0][br0 * LDSPAD + bs0 * 8]), bgsrc0);
  async_b128(lds_off(&sB[0][br1 * LDSPAD + bs1 * 8]), bgsrc1);

  int nk = K / 32;
  for (int i = 0; i < nk; ++i) {
    wait_async0();        // own transfers for buffer (i&1) complete
    __syncthreads();      // everyone's transfers visible; prev reads done
    if (i + 1 < nk) {     // overlap next tile's async copies with WMMAs
      int nb = (i + 1) & 1;
      int k = (i + 1) * 32;
      async_b128(lds_off(&sA[nb][arow * LDSPAD]) + alds, agsrc + k);
      async_b128(lds_off(&sB[nb][br0 * LDSPAD + bs0 * 8]), bgsrc0 + k);
      async_b128(lds_off(&sB[nb][br1 * LDSPAD + bs1 * 8]), bgsrc1 + k);
    }
    int buf = i & 1;
    const _Float16* pa = &sA[buf][0];
    const _Float16* pb = &sB[buf][0];
    v16h a0 = ld_a_rm(pa + (size_t)lrow * LDSPAD, LDSPAD, lane);
    v16h a1 = ld_a_rm(pa + (size_t)(lrow + 16) * LDSPAD, LDSPAD, lane);
    v16h b0 = ld_b(pb + (size_t)lcol * LDSPAD, LDSPAD, lane);
    v16h b1 = ld_b(pb + (size_t)(lcol + 16) * LDSPAD, LDSPAD, lane);
    c00 = wmma_f16(a0, b0, c00);
    c01 = wmma_f16(a0, b1, c01);
    c10 = wmma_f16(a1, b0, c10);
    c11 = wmma_f16(a1, b1, c11);
  }
  int row0 = rowblk + lrow, col0 = colblk + lcol;
  store_tile(c00, row0,      col0,      addsrc, C, C16, ldc, lane);
  store_tile(c01, row0,      col0 + 16, addsrc, C, C16, ldc, lane);
  store_tile(c10, row0 + 16, col0,      addsrc, C, C16, ldc, lane);
  store_tile(c11, row0 + 16, col0 + 16, addsrc, C, C16, ldc, lane);
}

// ---- flash attention: wave-per-(head, 16-row q tile), causal ---------------
__global__ void attn_kernel(const _Float16* __restrict__ Q,
                            const _Float16* __restrict__ K,
                            const _Float16* __restrict__ VT,
                            _Float16* __restrict__ AO) {
  __shared__ _Float16 plds[4 * 16 * 32];
  int lane = threadIdx.x & 31;
  int wv = threadIdx.x >> 5;
  int task = blockIdx.x * 4 + wv;
  int head = task >> 7;            // /128 q-tiles
  int qt   = task & 127;
  int kvh  = head >> 2;            // GQA 4:1
  int l = lane & 15, hf = lane >> 4;
  _Float16* pl = plds + wv * 512;

  v16h qa[4];
  const _Float16* qbase = Q + (size_t)(qt * 16) * DQ + head * HD_;
#pragma unroll
  for (int c = 0; c < 4; ++c) qa[c] = ld_a_rm(qbase + c * 32, DQ, lane);

  v8f o[8];
#pragma unroll
  for (int i = 0; i < 8; ++i) o[i] = zero8();
  float m[8], lsum[8];
#pragma unroll
  for (int r = 0; r < 8; ++r) { m[r] = -1e30f; lsum[r] = 0.f; }

  int qrow0 = qt * 16;
  int nkb = qt / 2 + 1;                       // causal: only needed k-blocks
  for (int kb = 0; kb < nkb; ++kb) {
    int kbase = kb * 32;
    v8f s0 = zero8(), s1 = zero8();
#pragma unroll
    for (int c = 0; c < 4; ++c) {             // K^T fragments == K row-major
      v16h b0 = ld_b(K + (size_t)kbase * DKV + kvh * HD_ + c * 32, DKV, lane);
      v16h b1 = ld_b(K + (size_t)(kbase + 16) * DKV + kvh * HD_ + c * 32, DKV, lane);
      s0 = wmma_f16(qa[c], b0, s0);
      s1 = wmma_f16(qa[c], b1, s1);
    }
    if (kb == nkb - 1) {                      // diagonal block: causal mask
#pragma unroll
      for (int r = 0; r < 8; ++r) {
        int row = qrow0 + r + hf * 8;
        if (kbase + l > row)       s0[r] += -1e9f;
        if (kbase + 16 + l > row)  s1[r] += -1e9f;
      }
    }
    float alpha[8];
    v8f p0, p1;
#pragma unroll
    for (int r = 0; r < 8; ++r) {             // online softmax row stats
      float mx = rmax16(fmaxf(s0[r], s1[r]));
      float mn = fmaxf(m[r], mx);
      alpha[r] = __expf(m[r] - mn);
      m[r] = mn;
      p0[r] = __expf(s0[r] - mn);
      p1[r] = __expf(s1[r] - mn);
      lsum[r] = lsum[r] * alpha[r] + rsum16(p0[r] + p1[r]);
    }
#pragma unroll
    for (int nt = 0; nt < 8; ++nt)
#pragma unroll
      for (int r = 0; r < 8; ++r) o[nt][r] *= alpha[r];
    // stage P (16x32 f16) through per-wave LDS; no block barrier (divergent waves)
#pragma unroll
    for (int r = 0; r < 8; ++r) {
      int row = r + hf * 8;
      pl[row * 32 + l]      = (_Float16)p0[r];
      pl[row * 32 + 16 + l] = (_Float16)p1[r];
    }
    asm volatile("s_wait_dscnt 0" ::: "memory");
    v16h pa = ld_a_rm(pl, 32, lane);
#pragma unroll
    for (int nt = 0; nt < 8; ++nt) {          // V^T fragments from vt[kv*HD+hd][t]
      v16h vb = ld_b(VT + (size_t)(kvh * HD_ + nt * 16) * T_ + kbase, T_, lane);
      o[nt] = wmma_f16(pa, vb, o[nt]);
    }
  }
  float inv[8];
#pragma unroll
  for (int r = 0; r < 8; ++r) inv[r] = 1.f / lsum[r];
#pragma unroll
  for (int nt = 0; nt < 8; ++nt)
#pragma unroll
    for (int r = 0; r < 8; ++r) {
      int row = qrow0 + r + hf * 8;
      AO[(size_t)row * DQ + head * HD_ + nt * 16 + l] = (_Float16)(o[nt][r] * inv[r]);
    }
}

// ---- gating + routing ------------------------------------------------------
__global__ void zero_cnt_kernel(int* c) { if (threadIdx.x < E_) c[threadIdx.x] = 0; }

__global__ void gate_route_kernel(const float* __restrict__ X,
                                  const float* __restrict__ LN2W,
                                  const float* __restrict__ GW,
                                  const float* __restrict__ BE,
                                  float* __restrict__ RW, int* __restrict__ CNT,
                                  int* __restrict__ LST) {
  int t = blockIdx.x, tid = threadIdx.x;
  const float* row = X + (size_t)t * H_;
  __shared__ float red[256];
  float s = 0.f;
  for (int i = tid; i < H_; i += 256) { float v = row[i]; s += v * v; }
  red[tid] = s; __syncthreads();
  for (int o = 128; o > 0; o >>= 1) { if (tid < o) red[tid] += red[tid + o]; __syncthreads(); }
  float rs = rsqrtf(red[0] / H_ + EPS_);
  __syncthreads();
  float acc[E_];
#pragma unroll
  for (int e = 0; e < E_; ++e) acc[e] = 0.f;
  for (int i = tid; i < H_; i += 256) {
    float xv = row[i] * rs * LN2W[i];
#pragma unroll
    for (int e = 0; e < E_; ++e) acc[e] += xv * GW[(size_t)i * E_ + e];
  }
  float lg[E_];
  for (int e = 0; e < E_; ++e) {
    red[tid] = acc[e]; __syncthreads();
    for (int o = 128; o > 0; o >>= 1) { if (tid < o) red[tid] += red[tid + o]; __syncthreads(); }
    lg[e] = red[0]; __syncthreads();
  }
  if (tid == 0) {
    float sig[E_], biased[E_];
#pragma unroll
    for (int e = 0; e < E_; ++e) { sig[e] = 1.f / (1.f + __expf(-lg[e])); biased[e] = sig[e] + BE[e]; }
    int i0 = 0;
    for (int e = 1; e < E_; ++e) if (biased[e] > biased[i0]) i0 = e;
    int i1 = (i0 == 0) ? 1 : 0;
    for (int e = 0; e < E_; ++e) { if (e == i0 || e == i1) continue; if (biased[e] > biased[i1]) i1 = e; }
    float r0 = sig[i0], r1 = sig[i1];
    float den = fmaxf(r0 + r1, 1e-12f);
    RW[t * 2] = r0 / den; RW[t * 2 + 1] = r1 / den;
    int p0 = atomicAdd(&CNT[i0], 1); LST[i0 * T_ + p0] = t * 2;
    int p1 = atomicAdd(&CNT[i1], 1); LST[i1 * T_ + p1] = t * 2 + 1;
  }
}

// ---- MoE stage 1: gu = silu(x@w1_e) * (x@w3_e), token-gathered -------------
__global__ void moe_stage1_kernel(const _Float16* __restrict__ X,
                                  const _Float16* __restrict__ W1T,
                                  const _Float16* __restrict__ W3T,
                                  const int* __restrict__ CNT,
                                  const int* __restrict__ LST,
                                  _Float16* __restrict__ GU) {
  int e = blockIdx.z, tt = blockIdx.y;
  int n = CNT[e];
  if (tt * 16 >= n) return;
  int lane = threadIdx.x & 31, wv = threadIdx.x >> 5;
  int col0 = blockIdx.x * 256 + wv * 32;
  const int* lp = LST + e * T_;
  int ep = tt * 16 + (lane & 15);
  int tok = (ep < n) ? (lp[ep] >> 1) : 0;
  const _Float16* arow = X + (size_t)tok * H_;
  const _Float16* w1b0 = W1T + (size_t)e * I_ * H_ + (size_t)col0 * H_;
  const _Float16* w1b1 = w1b0 + (size_t)16 * H_;
  const _Float16* w3b0 = W3T + (size_t)e * I_ * H_ + (size_t)col0 * H_;
  const _Float16* w3b1 = w3b0 + (size_t)16 * H_;
  v8f g0 = zero8(), g1 = zero8(), u0 = zero8(), u1 = zero8();
  for (int k = 0; k < H_; k += 32) {
    if (k + 32 < H_) {
      __builtin_prefetch(arow + (size_t)k + 32, 0, 3);
      __builtin_prefetch(w1b0 + (size_t)(lane & 15) * H_ + k + 32, 0, 3);
      __builtin_prefetch(w3b0 + (size_t)(lane & 15) * H_ + k + 32, 0, 3);
    }
    v16h a = ld_a(arow + k, lane);
    v16h b;
    b = ld_b(w1b0 + k, H_, lane); g0 = wmma_f16(a, b, g0);
    b = ld_b(w1b1 + k, H_, lane); g1 = wmma_f16(a, b, g1);
    b = ld_b(w3b0 + k, H_, lane); u0 = wmma_f16(a, b, u0);
    b = ld_b(w3b1 + k, H_, lane); u1 = wmma_f16(a, b, u1);
  }
  int l = lane & 15, hf = lane >> 4;
#pragma unroll
  for (int r = 0; r < 8; ++r) {
    int epr = tt * 16 + r + hf * 8;
    if (epr < n) {
      int idx2 = lp[epr];
      size_t base = (size_t)idx2 * I_ + col0 + l;
      float g = g0[r], u = u0[r];
      GU[base] = (_Float16)((g / (1.f + __expf(-g))) * u);
      g = g1[r]; u = u1[r];
      GU[base + 16] = (_Float16)((g / (1.f + __expf(-g))) * u);
    }
  }
}

// ---- MoE stage 2: y_slot = gu @ w2_e (token-gathered, scatter to slots) ----
__global__ void moe_stage2_kernel(const _Float16* __restrict__ GU,
                                  const _Float16* __restrict__ W2T,
                                  const int* __restrict__ CNT,
                                  const int* __restrict__ LST,
                                  float* __restrict__ Y) {
  int e = blockIdx.z, tt = blockIdx.y;
  int n = CNT[e];
  if (tt * 16 >= n) return;
  int lane = threadIdx.x & 31, wv = threadIdx.x >> 5;
  int col0 = blockIdx.x * 256 + wv * 32;
  const int* lp = LST + e * T_;
  int ep = tt * 16 + (lane & 15);
  int rowi = (ep < n) ? lp[ep] : 0;
  const _Float16* arow = GU + (size_t)rowi * I_;
  const _Float16* w2b0 = W2T + (size_t)e * H_ * I_ + (size_t)col0 * I_;
  const _Float16* w2b1 = w2b0 + (size_t)16 * I_;
  v8f c0 = zero8(), c1 = zero8();
  for (int k = 0; k < I_; k += 32) {
    if (k + 32 < I_) {
      __builtin_prefetch(arow + (size_t)k + 32, 0, 3);
      __builtin_prefetch(w2b0 + (size_t)(lane & 15) * I_ + k + 32, 0, 3);
    }
    v16h a = ld_a(arow + k, lane);
    v16h b0 = ld_b(w2b0 + k, I_, lane); c0 = wmma_f16(a, b0, c0);
    v16h b1 = ld_b(w2b1 + k, I_, lane); c1 = wmma_f16(a, b1, c1);
  }
  int l = lane & 15, hf = lane >> 4;
#pragma unroll
  for (int r = 0; r < 8; ++r) {
    int epr = tt * 16 + r + hf * 8;
    if (epr < n) {
      int idx2 = lp[epr];
      size_t base = (size_t)idx2 * H_ + col0 + l;
      Y[base] = c0[r];
      Y[base + 16] = c1[r];
    }
  }
}

// ---- final combine: out = x1 + w0*y_slot0 + w1*y_slot1 (fixed order) -------
__global__ void moe_combine_kernel(const float* __restrict__ X1,
                                   const float* __restrict__ Y,
                                   const float* __restrict__ RW,
                                   float* __restrict__ OUT) {
  size_t i = (size_t)blockIdx.x * 256 + threadIdx.x;
  int t = (int)(i >> 11);
  int hc = (int)(i & (H_ - 1));
  OUT[i] = X1[i] + RW[t * 2] * Y[((size_t)(t * 2)) * H_ + hc]
                 + RW[t * 2 + 1] * Y[((size_t)(t * 2 + 1)) * H_ + hc];
}

// ---- host side -------------------------------------------------------------
extern "C" void kernel_launch(void* const* d_in, const int* in_sizes, int n_in,
                              void* d_out, int out_size, void* d_ws, size_t ws_size,
                              hipStream_t stream) {
  const float* hidden = (const float*)d_in[0];
  // d_in[1] = attention_mask (causal triu(-1e9), implemented analytically)
  const float* cosb  = (const float*)d_in[2];
  const float* sinb  = (const float*)d_in[3];
  const float* ln1w  = (const float*)d_in[4];
  const float* ln2w  = (const float*)d_in[5];
  const float* wq    = (const float*)d_in[6];
  const float* wk    = (const float*)d_in[7];
  const float* wv    = (const float*)d_in[8];
  const float* wo    = (const float*)d_in[9];
  const float* qnw   = (const float*)d_in[10];
  const float* knw   = (const float*)d_in[11];
  const float* gatew = (const float*)d_in[12];
  const float* biase = (const float*)d_in[13];
  const float* w1    = (const float*)d_in[14];
  const float* w3    = (const float*)d_in[15];
  const float* w2    = (const float*)d_in[16];
  float* out = (float*)d_out;

  char* wsb = (char*)d_ws;
  size_t off = 0;
  auto alloc = [&](size_t bytes) -> void* {
    void* p = wsb + off;
    off = (off + bytes + 255) & ~(size_t)255;
    return p;
  };
  _Float16* wqT = (_Float16*)alloc((size_t)DQ * H_ * 2);
  _Float16* wkT = (_Float16*)alloc((size_t)DKV * H_ * 2);
  _Float16* wvT = (_Float16*)alloc((size_t)DKV * H_ * 2);
  _Float16* woT = (_Float16*)alloc((size_t)H_ * DQ * 2);
  _Float16* w1T = (_Float16*)alloc((size_t)E_ * I_ * H_ * 2);
  _Float16* w3T = (_Float16*)alloc((size_t)E_ * I_ * H_ * 2);
  _Float16* w2T = (_Float16*)alloc((size_t)E_ * H_ * I_ * 2);
  _Float16* xn1 = (_Float16*)alloc((size_t)T_ * H_ * 2);
  float* qraw   = (float*)alloc((size_t)T_ * DQ * 4);
  float* kraw   = (float*)alloc((size_t)T_ * DKV * 4);
  float* vraw   = (float*)alloc((size_t)T_ * DKV * 4);
  _Float16* q16 = (_Float16*)alloc((size_t)T_ * DQ * 2);
  _Float16* k16 = (_Float16*)alloc((size_t)T_ * DKV * 2);
  _Float16* vt16= (_Float16*)alloc((size_t)DKV * T_ * 2);
  _Float16* ao16= (_Float16*)alloc((size_t)T_ * DQ * 2);
  float* x1     = (float*)alloc((size_t)T_ * H_ * 4);
  _Float16* xn2 = (_Float16*)alloc((size_t)T_ * H_ * 2);
  float* rw     = (float*)alloc((size_t)T_ * 2 * 4);
  int* cnt      = (int*)alloc((size_t)E_ * 4);
  int* lst      = (int*)alloc((size_t)E_ * T_ * 4);
  _Float16* gu  = (_Float16*)alloc((size_t)T_ * 2 * I_ * 2);
  float* yslot  = (float*)alloc((size_t)T_ * 2 * H_ * 4);

  dim3 tb(32, 8);
  // weights -> f16 transposed ([N][K]) for B-fragment contiguity
  tr_f32_f16_kernel<<<dim3(DQ / 32, H_ / 32, 1), tb, 0, stream>>>(wq, wqT, H_, DQ);
  tr_f32_f16_kernel<<<dim3(DKV / 32, H_ / 32, 1), tb, 0, stream>>>(wk, wkT, H_, DKV);
  tr_f32_f16_kernel<<<dim3(DKV / 32, H_ / 32, 1), tb, 0, stream>>>(wv, wvT, H_, DKV);
  tr_f32_f16_kernel<<<dim3(H_ / 32, DQ / 32, 1), tb, 0, stream>>>(wo, woT, DQ, H_);
  tr_f32_f16_kernel<<<dim3(I_ / 32, H_ / 32, E_), tb, 0, stream>>>(w1, w1T, H_, I_);
  tr_f32_f16_kernel<<<dim3(I_ / 32, H_ / 32, E_), tb, 0, stream>>>(w3, w3T, H_, I_);
  tr_f32_f16_kernel<<<dim3(H_ / 32, I_ / 32, E_), tb, 0, stream>>>(w2, w2T, I_, H_);

  rmsnorm_kernel<<<T_, 256, 0, stream>>>(hidden, ln1w, xn1);

  gemm_f16_kernel<<<dim3(DQ / 128, T_ / 64), 256, 0, stream>>>(
      xn1, H_, wqT, H_, nullptr, qraw, nullptr, DQ, H_);
  gemm_f16_kernel<<<dim3(DKV / 128, T_ / 64), 256, 0, stream>>>(
      xn1, H_, wkT, H_, nullptr, kraw, nullptr, DKV, H_);
  gemm_f16_kernel<<<dim3(DKV / 128, T_ / 64), 256, 0, stream>>>(
      xn1, H_, wvT, H_, nullptr, vraw, nullptr, DKV, H_);

  qknorm_rope_kernel<<<T_, 256, 0, stream>>>(qraw, qnw, cosb, sinb, q16, DQ, SCALE_);
  qknorm_rope_kernel<<<T_, 256, 0, stream>>>(kraw, knw, cosb, sinb, k16, DKV, 1.0f);
  // V [T][DKV] f32 -> V^T [DKV][T] f16 (B-fragments for P@V)
  tr_f32_f16_kernel<<<dim3(DKV / 32, T_ / 32, 1), tb, 0, stream>>>(vraw, vt16, T_, DKV);

  attn_kernel<<<(NH_ * 128) / 4, 128, 0, stream>>>(q16, k16, vt16, ao16);

  gemm_f16_kernel<<<dim3(H_ / 128, T_ / 64), 256, 0, stream>>>(
      ao16, DQ, woT, DQ, hidden, x1, nullptr, H_, DQ);

  rmsnorm_kernel<<<T_, 256, 0, stream>>>(x1, ln2w, xn2);

  zero_cnt_kernel<<<1, 32, 0, stream>>>(cnt);
  gate_route_kernel<<<T_, 256, 0, stream>>>(x1, ln2w, gatew, biase, rw, cnt, lst);

  moe_stage1_kernel<<<dim3(I_ / 256, T_ / 16, E_), 256, 0, stream>>>(
      xn2, w1T, w3T, cnt, lst, gu);
  moe_stage2_kernel<<<dim3(H_ / 256, T_ / 16, E_), 256, 0, stream>>>(
      gu, w2T, cnt, lst, yslot);

  moe_combine_kernel<<<(T_ * H_) / 256, 256, 0, stream>>>(x1, yslot, rw, out);
}
